// ViTSelfAttention_10780367913577
// MI455X (gfx1250) — compile-verified, matrix operand order checked by reading, not run on hardware
//
#include <hip/hip_runtime.h>
#include <hip/hip_bf16.h>

// ---------------------------------------------------------------------------
// ViT self-attention block for MI455X (gfx1250).
//   0) x -> bf16 once; W{q,k,v,c} -> bf16 transposed [N][K] once
//   1) Q/K/V = (x @ W + b) [*0.125 for Q]  -- bf16 WMMA GEMM, TDM-staged tiles,
//      double-buffered LDS, outputs bf16
//   2) flash attention per (b,h), online softmax, bf16 output in [B,H,S,HD]
//      order (matches reference's reshape-without-transpose)
//   3) out = attn @ Wc + bc (f32 output)
// Fragments are aligned 128-bit LDS reads matching ISA 7.12.2 16-bit A-layout;
// B-operands are pre-transposed so fragments are contiguous.
// ---------------------------------------------------------------------------

typedef __attribute__((ext_vector_type(16))) __bf16          v16bf;
typedef __attribute__((ext_vector_type(8)))  float           v8f;
typedef __attribute__((ext_vector_type(8)))  unsigned short  ushort8;
typedef unsigned int u32x4 __attribute__((ext_vector_type(4)));
typedef int          i32x4 __attribute__((ext_vector_type(4)));
typedef int          i32x8 __attribute__((ext_vector_type(8)));

union BF16Frag { v16bf v; ushort8 h[2]; };

// f32 -> bf16, round-half-up
__device__ __forceinline__ unsigned short f2bf(float f) {
  union { float f; unsigned int u; } x; x.f = f;
  return (unsigned short)((x.u + 0x8000u) >> 16);
}
__device__ __forceinline__ unsigned int pk_bf2(float x, float y) {
  union { float f; unsigned int u; } a, b; a.f = x; b.f = y;
  return ((a.u + 0x8000u) >> 16) | ((b.u + 0x8000u) & 0xFFFF0000u);
}

// ---------------------------------------------------------------------------
// TDM: 2D tile (tileX x tileY) of 2-byte elements, row stride strideX elems,
// into LDS at lds_byte_addr with +padAmount DWORD padding every 2^padInterval
// 8-byte units (ISA 8.3/8.4 descriptor layout; group0.type=2, count=1).
// ---------------------------------------------------------------------------
__device__ __forceinline__ void tdm_load_tile_2d(
    const void* gsrc, unsigned lds_byte_addr,
    unsigned tileX, unsigned tileY,
    unsigned tensorX, unsigned tensorY, unsigned strideX,
    unsigned padInterval, unsigned padAmount) {
  const unsigned long long ga = (unsigned long long)(size_t)gsrc;
  u32x4 g0;
  g0[0] = 1u;                                       // count=1, user mode
  g0[1] = lds_byte_addr;                            // lds_addr
  g0[2] = (unsigned)ga;                             // global_addr[31:0]
  g0[3] = (unsigned)((ga >> 32) & 0x01FFFFFFull)    // global_addr[56:32]
          | (2u << 30);                             // type=2 ("image")
  i32x8 g1;
  g1[0] = (int)((1u << 16)                          // data_size = 2 bytes
              | (1u << 20)                          // pad_enable
              | (padInterval << 22) | (padAmount << 25));
  g1[1] = (int)((tensorX & 0xFFFFu) << 16);                         // dim0 lo16
  g1[2] = (int)(((tensorX >> 16) & 0xFFFFu) | ((tensorY & 0xFFFFu) << 16));
  g1[3] = (int)(((tensorY >> 16) & 0xFFFFu) | (tileX << 16));       // tile_dim0
  g1[4] = (int)(tileY & 0xFFFFu);                   // tile_dim1 (tile_dim2=0)
  g1[5] = (int)strideX;                             // tensor_dim0_stride lo32
  g1[6] = 0;
  g1[7] = 0;
  i32x4 gz = { 0, 0, 0, 0 };
#if __clang_major__ >= 23
  i32x8 gz8 = { 0, 0, 0, 0, 0, 0, 0, 0 };
  __builtin_amdgcn_tensor_load_to_lds(g0, g1, gz, gz, gz8, 0);
#else
  __builtin_amdgcn_tensor_load_to_lds(g0, g1, gz, gz, 0);
#endif
}

// ---------------------------------------------------------------------------
// one-time prep kernels
// ---------------------------------------------------------------------------
__global__ __launch_bounds__(256)
void convert_f32_bf16(const float* __restrict__ src, unsigned short* __restrict__ dst) {
  const size_t i = ((size_t)blockIdx.x * 256 + threadIdx.x) * 4;
  const float4 f = *(const float4*)(src + i);
  uint2 p; p.x = pk_bf2(f.x, f.y); p.y = pk_bf2(f.z, f.w);
  *(uint2*)(dst + i) = p;
}

// Wt[n][k] = bf16(W[k][n]); 32x32 LDS tile transpose, 256 threads
__global__ __launch_bounds__(256)
void transpose_f32_bf16(const float* __restrict__ W, unsigned short* __restrict__ Wt,
                        int K, int N) {
  __shared__ float t[32][33];
  const int tx = threadIdx.x & 31, ty = threadIdx.x >> 5;   // ty 0..7
  const int n0 = blockIdx.x * 32, k0 = blockIdx.y * 32;
#pragma unroll
  for (int i = 0; i < 32; i += 8)
    t[ty + i][tx] = W[(size_t)(k0 + ty + i) * N + n0 + tx];
  __syncthreads();
#pragma unroll
  for (int i = 0; i < 32; i += 8)
    Wt[(size_t)(n0 + ty + i) * K + k0 + tx] = f2bf(t[tx][ty + i]);
}

// ---------------------------------------------------------------------------
// GEMM: C[M,N] = (A[M,K] @ Bt[N,K]^T + bias) * oscale, A/Bt bf16, TDM staging.
// Block tile 128x128, BK=32, 256 threads = 8 waves, double-buffered LDS.
// ---------------------------------------------------------------------------
template <bool OUT_BF16>
__global__ __launch_bounds__(256)
void gemm_tdm_bf16wmma(const unsigned short* __restrict__ A,
                       const unsigned short* __restrict__ Bt,
                       const float* __restrict__ bias, void* __restrict__ Cout,
                       int M, int N, int K, float oscale) {
  __shared__ __align__(16) unsigned short lA [2][128][40];  // 80B rows (64B + 16B pad)
  __shared__ __align__(16) unsigned short lBt[2][128][40];

  const int tid  = threadIdx.x;
  const int lane = tid & 31;
  const int wave = tid >> 5;
  const int wm   = wave & 3;     // 0..3 -> 32-row strip
  const int wn   = wave >> 2;    // 0..1 -> 64-col strip
  const int half = lane >> 4;
  const int r    = lane & 15;
  const int ha   = half * 8;

  const int bm = blockIdx.x * 128;
  const int bn = blockIdx.y * 128;

  const unsigned short* aBase = A  + (size_t)bm * K;
  const unsigned short* bBase = Bt + (size_t)bn * K;
  const unsigned ldsA0 = (unsigned)(size_t)(void*)&lA [0][0][0];
  const unsigned ldsA1 = (unsigned)(size_t)(void*)&lA [1][0][0];
  const unsigned ldsB0 = (unsigned)(size_t)(void*)&lBt[0][0][0];
  const unsigned ldsB1 = (unsigned)(size_t)(void*)&lBt[1][0][0];

  const int nK = K / 32;
  if (wave == 0) {   // prologue: tiles k=0 (buf0) and k=1 (buf1)
    tdm_load_tile_2d(aBase,      ldsA0, 32, 128, K, M, K, 3, 3);
    tdm_load_tile_2d(bBase,      ldsB0, 32, 128, K, N, K, 3, 3);
    tdm_load_tile_2d(aBase + 32, ldsA1, 32, 128, K, M, K, 3, 3);
    tdm_load_tile_2d(bBase + 32, ldsB1, 32, 128, K, N, K, 3, 3);
  }

  v8f zero = {};
  v8f acc[2][4];
#pragma unroll
  for (int mi = 0; mi < 2; ++mi)
#pragma unroll
    for (int ni = 0; ni < 4; ++ni) acc[mi][ni] = zero;

  for (int k = 0; k < nK; ++k) {
    const int buf = k & 1;
    if (wave == 0) {              // tile k complete (TDM completes in order)
      if (k < nK - 1) __builtin_amdgcn_s_wait_tensorcnt(2);
      else            __builtin_amdgcn_s_wait_tensorcnt(0);
    }
    __syncthreads();

    // ---- fragments: two aligned b128 LDS loads each ----
    BF16Frag af[2], bf[4];
#pragma unroll
    for (int mi = 0; mi < 2; ++mi) {
      const int row = wm * 32 + mi * 16 + r;
      af[mi].h[0] = *(const ushort8*)&lA[buf][row][ha];
      af[mi].h[1] = *(const ushort8*)&lA[buf][row][16 + ha];
    }
#pragma unroll
    for (int ni = 0; ni < 4; ++ni) {
      const int col = wn * 64 + ni * 16 + r;
      bf[ni].h[0] = *(const ushort8*)&lBt[buf][col][ha];
      bf[ni].h[1] = *(const ushort8*)&lBt[buf][col][16 + ha];
    }

    // ---- 8 WMMA per wave per K-step ----
#pragma unroll
    for (int mi = 0; mi < 2; ++mi)
#pragma unroll
      for (int ni = 0; ni < 4; ++ni)
        acc[mi][ni] = __builtin_amdgcn_wmma_f32_16x16x32_bf16(
            false, af[mi].v, false, bf[ni].v, (short)0, acc[mi][ni], false, false);
    __syncthreads();   // all waves done with buf before TDM overwrites it

    if (wave == 0 && k + 2 < nK) {
      tdm_load_tile_2d(aBase + (size_t)(k + 2) * 32, buf ? ldsA1 : ldsA0,
                       32, 128, K, M, K, 3, 3);
      tdm_load_tile_2d(bBase + (size_t)(k + 2) * 32, buf ? ldsB1 : ldsB0,
                       32, 128, K, N, K, 3, 3);
    }
  }

  // ---- epilogue: C/D layout -> global, bias + scale ----
#pragma unroll
  for (int mi = 0; mi < 2; ++mi)
#pragma unroll
    for (int ni = 0; ni < 4; ++ni) {
      const int col = bn + wn * 64 + ni * 16 + r;
      const float bv = bias[col];
      const size_t rb = (size_t)(bm + wm * 32 + mi * 16 + 8 * half) * N + col;
#pragma unroll
      for (int rv = 0; rv < 8; ++rv) {
        const float val = (acc[mi][ni][rv] + bv) * oscale;
        if constexpr (OUT_BF16) ((unsigned short*)Cout)[rb + (size_t)rv * N] = f2bf(val);
        else                    ((float*)Cout)[rb + (size_t)rv * N] = val;
      }
    }
}

// ---------------------------------------------------------------------------
// Flash attention: one workgroup per (b, h, 64-row Q block); 4 waves,
// each wave owns 16 Q rows. KV tiles of 64, online softmax. Q/K/V bf16
// [B*S, D] (head h at column h*64, Q pre-scaled); output bf16 [B,H,S,HD].
// ---------------------------------------------------------------------------
#define ATT_HD 64

__global__ __launch_bounds__(128)
void flash_attn_bf16wmma(const unsigned short* __restrict__ Qm,
                         const unsigned short* __restrict__ Km,
                         const unsigned short* __restrict__ Vm,
                         unsigned short* __restrict__ O,
                         int Bn, int Hn, int Sn) {
  __shared__ __align__(16) unsigned short lK [64][72];      // [kv][d]
  __shared__ __align__(16) unsigned short lVt[64][72];      // [d][kv] transposed
  __shared__ __align__(16) unsigned short lP [4][16][72];   // per-wave P round-trip

  const int tid  = threadIdx.x;
  const int lane = tid & 31;
  const int wave = tid >> 5;
  const int half = lane >> 4;
  const int r    = lane & 15;
  const int ha   = half * 8;

  const int qb = blockIdx.x;
  const int h  = blockIdx.y;
  const int b  = blockIdx.z;

  const int    D     = Hn * ATT_HD;
  const size_t baseQ = (size_t)b * Sn * D + (size_t)h * ATT_HD;

  // staging: 16B (8 bf16) per thread per pass, 4 passes for a 64x64 tile
  const int e  = tid * 8;
  const int rw = e >> 6, cl = e & 63;    // +16 rows per pass

  // ---- Q tile -> LDS -> per-wave A-fragments (held in VGPRs all kernel) ----
  BF16Frag qf[2];
  {
    const unsigned short* qSrc = Qm + baseQ + (size_t)(qb * 64 + rw) * D + cl;
#pragma unroll
    for (int i = 0; i < 4; ++i)
      *(uint4*)&lK[rw + i * 16][cl] = *(const uint4*)(qSrc + (size_t)(i * 16) * D);
    __syncthreads();
#pragma unroll
    for (int kk = 0; kk < 2; ++kk) {
      qf[kk].h[0] = *(const ushort8*)&lK[wave * 16 + r][kk * 32 + ha];
      qf[kk].h[1] = *(const ushort8*)&lK[wave * 16 + r][kk * 32 + 16 + ha];
    }
    __syncthreads();
  }

  v8f zero = {};
  v8f oacc[4];
#pragma unroll
  for (int c = 0; c < 4; ++c) oacc[c] = zero;
  float m_i[8], l_i[8];
#pragma unroll
  for (int rv = 0; rv < 8; ++rv) { m_i[rv] = -3.402823466e38f; l_i[rv] = 0.0f; }

  const unsigned short* kSrc = Km + baseQ + (size_t)rw * D + cl;
  const unsigned short* vSrc = Vm + baseQ + (size_t)rw * D + cl;
  const size_t kvStep = (size_t)64 * D;

  const int nkv = Sn / 64;
  for (int j = 0; j < nkv; ++j) {
    // ---- stage K (row-major copy) and V (transposed scatter) ----
#pragma unroll
    for (int i = 0; i < 4; ++i) {
      const int row = rw + i * 16;
      *(uint4*)&lK[row][cl] = *(const uint4*)(kSrc + (size_t)(i * 16) * D);
      const uint4 v = *(const uint4*)(vSrc + (size_t)(i * 16) * D);
      lVt[cl + 0][row] = (unsigned short)(v.x);
      lVt[cl + 1][row] = (unsigned short)(v.x >> 16);
      lVt[cl + 2][row] = (unsigned short)(v.y);
      lVt[cl + 3][row] = (unsigned short)(v.y >> 16);
      lVt[cl + 4][row] = (unsigned short)(v.z);
      lVt[cl + 5][row] = (unsigned short)(v.z >> 16);
      lVt[cl + 6][row] = (unsigned short)(v.w);
      lVt[cl + 7][row] = (unsigned short)(v.w >> 16);
    }
    if (j + 1 < nkv) {
      __builtin_prefetch(kSrc + kvStep, 0, 1);
      __builtin_prefetch(vSrc + kvStep, 0, 1);
    }
    __syncthreads();

    // ---- scores: S = Q @ K^T (16 x 64 per wave) ----
    v8f sc[4];
#pragma unroll
    for (int c = 0; c < 4; ++c) {
      sc[c] = zero;
#pragma unroll
      for (int kk = 0; kk < 2; ++kk) {
        BF16Frag kf;
        kf.h[0] = *(const ushort8*)&lK[c * 16 + r][kk * 32 + ha];
        kf.h[1] = *(const ushort8*)&lK[c * 16 + r][kk * 32 + 16 + ha];
        sc[c] = __builtin_amdgcn_wmma_f32_16x16x32_bf16(
            false, qf[kk].v, false, kf.v, (short)0, sc[c], false, false);
      }
    }

    // ---- online softmax ----
    float ratio[8];
#pragma unroll
    for (int rv = 0; rv < 8; ++rv) {
      float t = sc[0][rv];
      t = fmaxf(t, sc[1][rv]); t = fmaxf(t, sc[2][rv]); t = fmaxf(t, sc[3][rv]);
#pragma unroll
      for (int mk = 8; mk >= 1; mk >>= 1)
        t = fmaxf(t, __shfl_xor(t, mk, 32));
      const float mn = fmaxf(m_i[rv], t);
      ratio[rv] = __expf(m_i[rv] - mn);
      m_i[rv] = mn;
    }
#pragma unroll
    for (int c = 0; c < 4; ++c)
#pragma unroll
      for (int rv = 0; rv < 8; ++rv)
        sc[c][rv] = __expf(sc[c][rv] - m_i[rv]);
#pragma unroll
    for (int rv = 0; rv < 8; ++rv) {
      float t = sc[0][rv] + sc[1][rv] + sc[2][rv] + sc[3][rv];
#pragma unroll
      for (int mk = 8; mk >= 1; mk >>= 1)
        t += __shfl_xor(t, mk, 32);
      l_i[rv] = l_i[rv] * ratio[rv] + t;
    }
#pragma unroll
    for (int c = 0; c < 4; ++c)
#pragma unroll
      for (int rv = 0; rv < 8; ++rv)
        oacc[c][rv] *= ratio[rv];

    // ---- P: C-layout -> LDS -> A-layout (per-wave region) ----
#pragma unroll
    for (int c = 0; c < 4; ++c)
#pragma unroll
      for (int rv = 0; rv < 8; ++rv)
        lP[wave][rv + 8 * half][c * 16 + r] = f2bf(sc[c][rv]);

    BF16Frag pf[2];
#pragma unroll
    for (int kk = 0; kk < 2; ++kk) {
      pf[kk].h[0] = *(const ushort8*)&lP[wave][r][kk * 32 + ha];
      pf[kk].h[1] = *(const ushort8*)&lP[wave][r][kk * 32 + 16 + ha];
    }

    // ---- O += P @ V ----
#pragma unroll
    for (int c = 0; c < 4; ++c)
#pragma unroll
      for (int kk = 0; kk < 2; ++kk) {
        BF16Frag vf;
        vf.h[0] = *(const ushort8*)&lVt[c * 16 + r][kk * 32 + ha];
        vf.h[1] = *(const ushort8*)&lVt[c * 16 + r][kk * 32 + 16 + ha];
        oacc[c] = __builtin_amdgcn_wmma_f32_16x16x32_bf16(
            false, pf[kk].v, false, vf.v, (short)0, oacc[c], false, false);
      }
    __syncthreads();

    kSrc += kvStep;
    vSrc += kvStep;
  }

  // ---- finalize: O / l -> bf16, [B,H,S,HD]-contiguous ----
  unsigned short* op =
      O + ((size_t)(b * Hn + h) * Sn + qb * 64 + wave * 16 + 8 * half) * ATT_HD;
#pragma unroll
  for (int c = 0; c < 4; ++c) {
    const int d = c * 16 + r;
#pragma unroll
    for (int rv = 0; rv < 8; ++rv)
      op[(size_t)rv * ATT_HD + d] = f2bf(oacc[c][rv] / l_i[rv]);
  }
}

// ---------------------------------------------------------------------------
extern "C" void kernel_launch(void* const* d_in, const int* in_sizes, int n_in,
                              void* d_out, int out_size, void* d_ws, size_t ws_size,
                              hipStream_t stream) {
  (void)in_sizes; (void)n_in; (void)out_size; (void)ws_size;

  const float* x  = (const float*)d_in[0];
  const float* Wq = (const float*)d_in[1];
  const float* bq = (const float*)d_in[2];
  const float* Wk = (const float*)d_in[3];
  const float* bk = (const float*)d_in[4];
  const float* Wv = (const float*)d_in[5];
  const float* bv = (const float*)d_in[6];
  const float* Wc = (const float*)d_in[7];
  const float* bc = (const float*)d_in[8];
  float* out = (float*)d_out;

  const int Bn = 16, S = 1024, D = 768, H = 12;
  const size_t tok = (size_t)Bn * S;   // 16384 rows
  const size_t mat = tok * D;          // elements per [B*S, D] buffer
  const size_t wsz = (size_t)D * D;

  unsigned short* xb  = (unsigned short*)d_ws;   // bf16 buffers
  unsigned short* Wqt = xb + mat;
  unsigned short* Wkt = Wqt + wsz;
  unsigned short* Wvt = Wkt + wsz;
  unsigned short* Wct = Wvt + wsz;
  unsigned short* Qb  = Wct + wsz;
  unsigned short* Kb  = Qb + mat;
  unsigned short* Vb  = Kb + mat;
  unsigned short* Ab  = Vb + mat;      // attention out, [B,H,S,HD] contiguous

  // phase 0: one-time converts / transposes
  convert_f32_bf16<<<dim3((unsigned)(mat / 1024)), dim3(256), 0, stream>>>(x, xb);
  dim3 gT((unsigned)(D / 32), (unsigned)(D / 32)), bT(256);
  transpose_f32_bf16<<<gT, bT, 0, stream>>>(Wq, Wqt, D, D);
  transpose_f32_bf16<<<gT, bT, 0, stream>>>(Wk, Wkt, D, D);
  transpose_f32_bf16<<<gT, bT, 0, stream>>>(Wv, Wvt, D, D);
  transpose_f32_bf16<<<gT, bT, 0, stream>>>(Wc, Wct, D, D);

  // phase 1: QKV projections (Q pre-scaled by 1/sqrt(HD))
  dim3 gG((unsigned)(tok / 128), (unsigned)(D / 128)), bG(256);
  gemm_tdm_bf16wmma<true><<<gG, bG, 0, stream>>>(xb, Wqt, bq, Qb, (int)tok, D, D, 0.125f);
  gemm_tdm_bf16wmma<true><<<gG, bG, 0, stream>>>(xb, Wkt, bk, Kb, (int)tok, D, D, 1.0f);
  gemm_tdm_bf16wmma<true><<<gG, bG, 0, stream>>>(xb, Wvt, bv, Vb, (int)tok, D, D, 1.0f);

  // phase 2: attention
  dim3 gA((unsigned)(S / 64), (unsigned)H, (unsigned)Bn), bA(128);
  flash_attn_bf16wmma<<<gA, bA, 0, stream>>>(Qb, Kb, Vb, Ab, Bn, H, S);

  // phase 3: output projection (f32 out)
  gemm_tdm_bf16wmma<false><<<gG, bG, 0, stream>>>(Ab, Wct, bc, out, (int)tok, D, D, 1.0f);
}
